// NodeEmbedder_87935160418864
// MI455X (gfx1250) — matrix operand order repaired and checked by reading, not compiled, not run on hardware
//
#include <hip/hip_runtime.h>
#include <hip/hip_bf16.h>
#include <math.h>

// ---------------------------------------------------------------------------
// Stacked LSTM (L=2, H=256, T=64, N=4096) for MI455X (gfx1250, wave32, WMMA).
//   * bf16 WMMA (v_wmma_f32_16x16x32_bf16), f32 accumulation.
//   * Each workgroup owns NT=32 nodes; recurrence is fully private to the WG.
//   * Combined GEMM per layer/step: [NT,512] x [512,1024], A = [x | h].
//   * Weights repacked once into per-lane WMMA B-fragment layout (L2-resident).
//   * Wave grid 1x16: each wave owns 4 unique col-tiles and BOTH row blocks,
//     so every B fragment is loaded exactly once per WG (1 global_load_b128
//     per WMMA instead of 2).
//   * c-state in registers; h-state in LDS (bf16); z staged in LDS (f32).
//   * global_prefetch of next timestep's embedding rows.
// ---------------------------------------------------------------------------

typedef __attribute__((ext_vector_type(16))) __bf16 v16bf;
typedef __attribute__((ext_vector_type(8)))  __bf16 v8bf;
typedef __attribute__((ext_vector_type(8)))  float  v8f;

#define H      256
#define G      1024        // 4*H
#define TSTEPS 64
#define NT     32          // nodes per workgroup
#define NTHREADS 512       // 16 waves
#define KC     16          // K chunks: 512 / 32

// ---------------------------------------------------------------------------
// Weight repack: [Wx;Wh] (f32, [2][256][1024] each) -> bf16 WMMA B fragments.
// Combined K = 512 (rows 0..255 = Wx, 256..511 = Wh).
// Packed layout: packed[l][kc][nt][lane][s], s = 0..15 bf16 (32B per lane).
// Dense 16-bit B 32x16 layout: lane = column (N = nt*16 + lane%16);
//   lanes 0-15 hold K = kc*32 + {0..7,16..23}, lanes 16-31 hold K = kc*32 + {8..15,24..31}.
// ---------------------------------------------------------------------------
__global__ void lstm_pack_weights(const float* __restrict__ Wx,
                                  const float* __restrict__ Wh,
                                  __bf16* __restrict__ packed) {
    int idx = blockIdx.x * 256 + threadIdx.x;      // 2*16*64*32*16 = 1048576
    int s    =  idx        & 15;
    int lane = (idx >> 4)  & 31;
    int nt   = (idx >> 9)  & 63;
    int kc   = (idx >> 15) & 15;
    int l    = (idx >> 19) & 1;

    int klocal = (s < 8) ? ((lane >> 4) * 8 + s) : (16 + (lane >> 4) * 8 + (s - 8));
    int k = kc * 32 + klocal;                      // 0..511 combined K
    int n = nt * 16 + (lane & 15);                 // 0..1023 output column

    float v = (k < H) ? Wx[((size_t)l * H + k) * G + n]
                      : Wh[((size_t)l * H + (k - H)) * G + n];
    packed[idx] = (__bf16)v;
}

// ---------------------------------------------------------------------------
// Main persistent-per-tile LSTM kernel. grid = N/NT = 128 blocks x 512 thr.
// ---------------------------------------------------------------------------
__global__ __launch_bounds__(NTHREADS, 1)
void lstm_stacked_kernel(const int*   __restrict__ data,    // [N, T] tokens
                         const float* __restrict__ embed,   // [V, H]
                         const float* __restrict__ bias,    // [2, 4H]
                         const __bf16* __restrict__ packedB,// packed weights
                         float*       __restrict__ out) {   // [N, H]
    __shared__ __align__(16) __bf16 x_lds[NT * H];          // 16 KB
    __shared__ __align__(16) __bf16 h_lds[2][NT * H];       // 32 KB
    __shared__ __align__(16) float  z_lds[NT * G];          // 128 KB

    const int tid  = threadIdx.x;
    const int wave = tid >> 5;
    const int lane = tid & 31;
    const int lm   = lane & 15;       // sub-tile row/col selector
    const int lh   = lane >> 4;       // lane half
    const int nodeBase = blockIdx.x * NT;

    // Wave grid 1x16: wave wc owns cols [wc*64, wc*64+64) for BOTH row blocks.
    const int wc = wave;

    // Gate-combine ownership: thread -> (node gn, 16 cols starting at gj).
    const int gn = tid >> 4;
    const int gj = (tid & 15) << 4;

    // Cell state entirely in registers: 2 layers x 16 elements.
    float c_reg[2][16];
#pragma unroll
    for (int l = 0; l < 2; ++l)
#pragma unroll
        for (int k = 0; k < 16; ++k) c_reg[l][k] = 0.0f;

    // h starts at zero.
    for (int i = tid; i < 2 * NT * H; i += NTHREADS)
        (&h_lds[0][0])[i] = (__bf16)0.0f;
    __syncthreads();

    for (int t = 0; t < TSTEPS; ++t) {
        // ---- Fused embedding gather: x_t = bf16(embed[data[n, t]]) --------
        {
            int n  = tid >> 4;
            int c0 = (tid & 15) << 4;
            int tok = data[(size_t)(nodeBase + n) * TSTEPS + t];
            const float4* src = (const float4*)(embed + (size_t)tok * H + c0);
#pragma unroll
            for (int q = 0; q < 4; ++q) {
                float4 f = src[q];
                __bf16* dst = &x_lds[n * H + c0 + q * 4];
                dst[0] = (__bf16)f.x; dst[1] = (__bf16)f.y;
                dst[2] = (__bf16)f.z; dst[3] = (__bf16)f.w;
            }
            // Prefetch next timestep's embedding rows (global_prefetch_b8).
            if (t + 1 < TSTEPS) {
                int tok2 = data[(size_t)(nodeBase + n) * TSTEPS + t + 1];
                __builtin_prefetch(embed + (size_t)tok2 * H + c0, 0, 0);
            }
        }
        __syncthreads();

        for (int l = 0; l < 2; ++l) {
            const __bf16* inp = (l == 0) ? &x_lds[0] : &h_lds[0][0];
            const __bf16* hp  = &h_lds[l][0];
            const __bf16* Bb  = packedB + (size_t)l * (KC * 64 * 512);

            // ---- GEMM: z[32,1024] = [inp | h] (K=512) x packedB ----------
            v8f acc[2][4] = {};

#pragma unroll 4
            for (int kc = 0; kc < KC; ++kc) {
                // A fragments for both 16-row blocks (documented interleaved
                // layout): lanes 0-15 K=koff+{0..7,16..23}; lanes 16-31: +8.
                const __bf16* asrc = (kc < 8) ? inp : hp;
                const int koff = (kc & 7) * 32;
                union { v16bf v; v8bf h2[2]; } afr[2];
#pragma unroll
                for (int r = 0; r < 2; ++r) {
                    const __bf16* ap =
                        asrc + (r * 16 + lm) * H + koff + lh * 8;
                    afr[r].h2[0] = *(const v8bf*)(ap);      // ds_load_b128
                    afr[r].h2[1] = *(const v8bf*)(ap + 16); // ds_load_b128
                }

#pragma unroll
                for (int ct = 0; ct < 4; ++ct) {
                    int nt = wc * 4 + ct;
                    const __bf16* bp =
                        Bb + ((size_t)(kc * 64 + nt) * 512) + lane * 16;
                    union { v16bf v; v8bf h2[2]; } bfr;
                    bfr.h2[0] = *(const v8bf*)(bp);      // global_load_b128
                    bfr.h2[1] = *(const v8bf*)(bp + 8);  // global_load_b128
                    // Reuse the B fragment for both row blocks.
                    acc[0][ct] = __builtin_amdgcn_wmma_f32_16x16x32_bf16(
                        false, afr[0].v, false, bfr.v,
                        (short)0, acc[0][ct], false, false);
                    acc[1][ct] = __builtin_amdgcn_wmma_f32_16x16x32_bf16(
                        false, afr[1].v, false, bfr.v,
                        (short)0, acc[1][ct], false, false);
                }
            }

            // ---- Spill z tiles to LDS (C/D layout: col=lane%16, M=v+8*lh)
#pragma unroll
            for (int r = 0; r < 2; ++r) {
#pragma unroll
                for (int ct = 0; ct < 4; ++ct) {
                    int col = wc * 64 + ct * 16 + lm;
#pragma unroll
                    for (int v = 0; v < 8; ++v) {
                        int row = r * 16 + lh * 8 + v;
                        z_lds[row * G + col] = acc[r][ct][v];
                    }
                }
            }
            __syncthreads();

            // ---- Gate combine (i,f,g,o), c in registers -------------------
            {
                const float* brow = bias + l * G;
                const float* zrow = &z_lds[gn * G];
#pragma unroll
                for (int k = 0; k < 16; ++k) {
                    int j = gj + k;
                    float zi = zrow[j      ] + brow[j      ];
                    float zf = zrow[j + 256] + brow[j + 256];
                    float zg = zrow[j + 512] + brow[j + 512];
                    float zo = zrow[j + 768] + brow[j + 768];
                    float ig = 1.0f / (1.0f + __expf(-zi));
                    float fg = 1.0f / (1.0f + __expf(-zf));
                    float og = 1.0f / (1.0f + __expf(-zo));
                    float gg = tanhf(zg);
                    float c  = fg * c_reg[l][k] + ig * gg;
                    c_reg[l][k] = c;
                    float h  = og * tanhf(c);
                    h_lds[l][gn * H + j] = (__bf16)h;
                    if (l == 1 && t == TSTEPS - 1)
                        out[(size_t)(nodeBase + gn) * H + j] = h;
                }
            }
            __syncthreads();
        }
    }
}

// ---------------------------------------------------------------------------
// Harness entry point. Inputs: data, embed, Wx, Wh, b. Output: [N, H] f32.
// ---------------------------------------------------------------------------
extern "C" void kernel_launch(void* const* d_in, const int* in_sizes, int n_in,
                              void* d_out, int out_size, void* d_ws, size_t ws_size,
                              hipStream_t stream) {
    const int*   data  = (const int*)  d_in[0];
    const float* embed = (const float*)d_in[1];
    const float* Wx    = (const float*)d_in[2];
    const float* Wh    = (const float*)d_in[3];
    const float* b     = (const float*)d_in[4];
    float* out = (float*)d_out;

    // Workspace: packed bf16 weights, 2 * 16 * 64 * 32 * 16 * 2B = 2 MB.
    __bf16* packed = (__bf16*)d_ws;

    // One-shot (deterministic, graph-safe) weight repack into WMMA fragments.
    lstm_pack_weights<<<4096, 256, 0, stream>>>(Wx, Wh, packed);

    // 4096 nodes / 32 per workgroup = 128 workgroups, 512 threads each.
    lstm_stacked_kernel<<<128, NTHREADS, 0, stream>>>(data, embed, b, packed, out);
}
